// BiLSTM_CRF_83811991814363
// MI455X (gfx1250) — matrix operand order repaired and checked by reading, not compiled, not run on hardware
//
#include <hip/hip_runtime.h>
#include <hip/hip_bf16.h>
#include <stdint.h>

// Problem constants (match reference)
#define BB  256   // batch
#define SS  512   // sequence length
#define EE  256   // embedding dim
#define HH  256   // hidden per direction
#define TT  20    // tags

typedef __attribute__((ext_vector_type(16))) __bf16 v16bf_t;
typedef __attribute__((ext_vector_type(8)))  __bf16 v8bf_t;
typedef __attribute__((ext_vector_type(8)))  float  v8f_t;

__device__ __forceinline__ float sigf(float x) { return 1.0f / (1.0f + __expf(-x)); }
__device__ __forceinline__ float fast_tanh(float x) { return 2.0f / (1.0f + __expf(-2.0f * x)) - 1.0f; }

// ---------------------------------------------------------------------------
// Utility kernels
// ---------------------------------------------------------------------------
__global__ __launch_bounds__(256)
void zero_kernel(uint32_t* __restrict__ p, size_t ndw) {
  size_t i = (size_t)blockIdx.x * blockDim.x + threadIdx.x;
  size_t stride = (size_t)gridDim.x * blockDim.x;
  for (; i < ndw; i += stride) p[i] = 0u;
}

__global__ __launch_bounds__(256)
void cvt_bf16_kernel(const float* __restrict__ src, __hip_bfloat16* __restrict__ dst, int n) {
  int i = blockIdx.x * blockDim.x + threadIdx.x;
  if (i < n) dst[i] = __float2bfloat16(src[i]);
}

// Embedding gather: xe[s][b][e] = embed[sentences[b][s]][e]   (row 0 -> zeros)
__global__ __launch_bounds__(256)
void embed_kernel(const int* __restrict__ sent, const float* __restrict__ emb,
                  __hip_bfloat16* __restrict__ xe) {
  const size_t idx = (size_t)blockIdx.x * blockDim.x + threadIdx.x;  // one thread per 8 elems
  const size_t total = (size_t)SS * BB * (EE / 8);
  if (idx >= total) return;
  const int e8 = (int)(idx & (EE / 8 - 1));
  const size_t sb = idx >> 5;            // s*BB + b
  const int s = (int)(sb / BB);
  const int b = (int)(sb % BB);
  const int tok = sent[(size_t)b * SS + s];
  __hip_bfloat16* dst = xe + sb * EE + (size_t)e8 * 8;
  if (tok == 0) {
    const __hip_bfloat16 z = __float2bfloat16(0.0f);
#pragma unroll
    for (int j = 0; j < 8; ++j) dst[j] = z;
  } else {
    const float* src = emb + (size_t)tok * EE + (size_t)e8 * 8;
#pragma unroll
    for (int j = 0; j < 8; ++j) dst[j] = __float2bfloat16(src[j]);
  }
}

// ---------------------------------------------------------------------------
// One LSTM timestep, both directions, WMMA bf16 GEMM [B x (IN_DIM+H)] * [.. x 4H]
// grid: 64 blocks x 256 thr = 512 waves = 2 dirs * 16 m-tiles * 16 h-col tiles.
// Each wave computes a 16x16 tile of each of the 4 gates (i/f/g/o) so the
// cell update can be applied in-registers from the WMMA C/D layout.
// ---------------------------------------------------------------------------
template <int IN_DIM>
__global__ __launch_bounds__(256)
void lstm_step_kernel(const __hip_bfloat16* __restrict__ xin,   // [SS][BB][IN_DIM]
                      const __hip_bfloat16* __restrict__ wih_f, // [4H][IN_DIM]
                      const __hip_bfloat16* __restrict__ whh_f, // [4H][H]
                      const __hip_bfloat16* __restrict__ wih_b,
                      const __hip_bfloat16* __restrict__ whh_b,
                      const float* __restrict__ bih_f, const float* __restrict__ bhh_f,
                      const float* __restrict__ bih_b, const float* __restrict__ bhh_b,
                      __hip_bfloat16* __restrict__ hstate,      // [dir][2][BB][HH] double-buffered
                      float* __restrict__ cstate,               // [dir][BB][HH]
                      __hip_bfloat16* __restrict__ hout,        // [SS][BB][2*HH]
                      int t) {
  const int wave = (int)((blockIdx.x * blockDim.x + threadIdx.x) >> 5);
  const int lane = (int)(threadIdx.x & 31u);
  const int dir  = wave >> 8;        // 0 = fwd, 1 = bwd
  const int rem  = wave & 255;
  const int mt   = rem >> 4;         // batch tile 0..15
  const int ng   = rem & 15;         // hidden-column tile 0..15
  const int tt   = dir ? (SS - 1 - t) : t;

  const __hip_bfloat16* wih = dir ? wih_b : wih_f;
  const __hip_bfloat16* whh = dir ? whh_b : whh_f;
  const float* bih = dir ? bih_b : bih_f;
  const float* bhh = dir ? bhh_b : bhh_f;

  const __hip_bfloat16* hprev = hstate + (size_t)(dir * 2 + (t & 1)) * (BB * HH);
  __hip_bfloat16*       hnext = hstate + (size_t)(dir * 2 + ((t + 1) & 1)) * (BB * HH);
  float* cst = cstate + (size_t)dir * (BB * HH);

  const int half  = lane >> 4;       // which 16-lane half
  const int r     = lane & 15;
  const int arow  = mt * 16 + r;     // A-matrix row (batch index)
  const int akoff = half ? 16 : 0;   // A fragment K sub-offset (contiguous, matches B)
  const int bkoff = half ? 16 : 0;   // B fragment K sub-offset

  const __bf16* xrow = reinterpret_cast<const __bf16*>(xin + ((size_t)tt * BB + arow) * IN_DIM);
  const __bf16* hrow = reinterpret_cast<const __bf16*>(hprev + (size_t)arow * HH);
  const __bf16* wihp = reinterpret_cast<const __bf16*>(wih);
  const __bf16* whhp = reinterpret_cast<const __bf16*>(whh);

  v8f_t acc[4] = {};

  constexpr int KT = IN_DIM + HH;    // fused K: [x_t ; h_{t-1}]
  for (int k0 = 0; k0 < KT; k0 += 32) {
    const int ka = k0 + akoff;
    v8bf_t alo, ahi;
    if (k0 < IN_DIM) {
      alo = *reinterpret_cast<const v8bf_t*>(xrow + ka);
      ahi = *reinterpret_cast<const v8bf_t*>(xrow + ka + 8);
    } else {
      alo = *reinterpret_cast<const v8bf_t*>(hrow + (ka - IN_DIM));
      ahi = *reinterpret_cast<const v8bf_t*>(hrow + (ka - IN_DIM) + 8);
    }
    v16bf_t afrag = __builtin_shufflevector(alo, ahi,
        0, 1, 2, 3, 4, 5, 6, 7, 8, 9, 10, 11, 12, 13, 14, 15);
#pragma unroll
    for (int g = 0; g < 4; ++g) {               // gates i, f, g, o
      const int n = g * 256 + ng * 16 + r;      // output gate column (B-matrix N)
      v16bf_t bfrag;
      if (k0 < IN_DIM) {
        bfrag = *reinterpret_cast<const v16bf_t*>(wihp + (size_t)n * IN_DIM + k0 + bkoff);
      } else {
        bfrag = *reinterpret_cast<const v16bf_t*>(whhp + (size_t)n * HH + (k0 - IN_DIM) + bkoff);
      }
      acc[g] = __builtin_amdgcn_wmma_f32_16x16x32_bf16(
          false, afrag, false, bfrag, (short)0, acc[g], false, false);
    }
  }

  // Gate combine + cell update from WMMA C/D layout:
  // lane -> N column (hidden unit), VGPR v -> M row (batch), +8 for upper lanes.
  const int hcol = ng * 16 + r;
  const float bi = bih[hcol]       + bhh[hcol];
  const float bf = bih[256 + hcol] + bhh[256 + hcol];
  const float bg = bih[512 + hcol] + bhh[512 + hcol];
  const float bo = bih[768 + hcol] + bhh[768 + hcol];
  const int rbase = mt * 16 + (half ? 8 : 0);

#pragma unroll
  for (int v = 0; v < 8; ++v) {
    const int brow = rbase + v;
    const float gi = sigf(acc[0][v] + bi);
    const float gf = sigf(acc[1][v] + bf);
    const float gg = fast_tanh(acc[2][v] + bg);
    const float go = sigf(acc[3][v] + bo);
    const size_t ci = (size_t)brow * HH + hcol;
    const float cn = gf * cst[ci] + gi * gg;
    cst[ci] = cn;
    const float hn = go * fast_tanh(cn);
    const __hip_bfloat16 hb = __float2bfloat16(hn);
    hnext[ci] = hb;
    hout[((size_t)tt * BB + brow) * (2 * HH) + (size_t)dir * HH + hcol] = hb;
  }
}

// ---------------------------------------------------------------------------
// Emissions: em[s][b][tag] = h2[s][b][:] . W_out[tag][:] + b_out[tag]
// one wave per (s,b); lane = tag (20 active of 32)
// ---------------------------------------------------------------------------
__global__ __launch_bounds__(256)
void emissions_kernel(const __hip_bfloat16* __restrict__ h2, const float* __restrict__ Wout,
                      const float* __restrict__ bout, float* __restrict__ em) {
  const int wave = (int)((blockIdx.x * blockDim.x + threadIdx.x) >> 5);  // = s*BB+b
  const int lane = (int)(threadIdx.x & 31u);
  if (wave >= SS * BB) return;
  const __hip_bfloat16* hr = h2 + (size_t)wave * (2 * HH);
  if (lane < TT) {
    const float* wr = Wout + (size_t)lane * (2 * HH);
    float acc = 0.0f;
    for (int k = 0; k < 2 * HH; ++k) acc += __bfloat162float(hr[k]) * wr[k];
    em[(size_t)wave * TT + lane] = acc + bout[lane];
  }
}

// Gold path score per batch item
__global__ __launch_bounds__(256)
void gold_kernel(const int* __restrict__ labels, const float* __restrict__ em,
                 const float* __restrict__ trans, float* __restrict__ score) {
  const int b = blockIdx.x * blockDim.x + threadIdx.x;
  if (b >= BB) return;
  float sc = 0.0f;
  int prev = 0;
  for (int s = 0; s < SS; ++s) {
    const int tag = labels[(size_t)b * SS + s];
    if (tag != 0) sc += em[((size_t)s * BB + b) * TT + tag];
    if (s > 0 && tag != 0 && prev != 0) sc += trans[prev * TT + tag];
    prev = tag;
  }
  score[b] = sc;
}

// CRF forward (partition) — wave per batch item, lane = tag, shfl-logsumexp
__global__ __launch_bounds__(256)
void crf_forward_kernel(const float* __restrict__ em, const float* __restrict__ trans,
                        float* __restrict__ part) {
  const int wave = (int)((blockIdx.x * blockDim.x + threadIdx.x) >> 5);
  const int lane = (int)(threadIdx.x & 31u);
  if (wave >= BB) return;
  const int b = wave;
  const int a = (lane < TT) ? lane : 0;  // clamp inactive lanes to a valid row
  float tr[TT];
#pragma unroll
  for (int c = 0; c < TT; ++c) tr[c] = trans[a * TT + c];
  float fv = (lane == 0) ? 0.0f : -10000.0f;
  for (int s = 0; s < SS; ++s) {
    float vals[TT];
    float m = -3.4e38f;
#pragma unroll
    for (int c = 0; c < TT; ++c) {
      const float v = __shfl(fv, c, 32) + tr[c];
      vals[c] = v;
      m = fmaxf(m, v);
    }
    float ssum = 0.0f;
#pragma unroll
    for (int c = 0; c < TT; ++c) ssum += __expf(vals[c] - m);
    fv = m + __logf(ssum) + em[((size_t)s * BB + b) * TT + a];
  }
  float m = -3.4e38f;
#pragma unroll
  for (int c = 0; c < TT; ++c) m = fmaxf(m, __shfl(fv, c, 32));
  float ssum = 0.0f;
#pragma unroll
  for (int c = 0; c < TT; ++c) ssum += __expf(__shfl(fv, c, 32) - m);
  if (lane == 0) part[b] = m + __logf(ssum);
}

// loss = mean(partition - score)
__global__ __launch_bounds__(256)
void loss_kernel(const float* __restrict__ part, const float* __restrict__ score,
                 float* __restrict__ out) {
  __shared__ float red[BB];
  const int i = threadIdx.x;
  red[i] = part[i] - score[i];
  __syncthreads();
  for (int st = BB / 2; st > 0; st >>= 1) {
    if (i < st) red[i] += red[i + st];
    __syncthreads();
  }
  if (i == 0) out[0] = red[0] / (float)BB;
}

// ---------------------------------------------------------------------------
// Host launcher
// ---------------------------------------------------------------------------
extern "C" void kernel_launch(void* const* d_in, const int* in_sizes, int n_in,
                              void* d_out, int out_size, void* d_ws, size_t ws_size,
                              hipStream_t stream) {
  (void)in_sizes; (void)n_in; (void)out_size; (void)ws_size;
  typedef __hip_bfloat16 bf16;

  // Inputs in setup_inputs() dict order, params flattened in insertion order.
  const int*   sent   = (const int*)d_in[0];
  const int*   labels = (const int*)d_in[1];
  const float* embedw = (const float*)d_in[2];
  const float* l1f_wih = (const float*)d_in[3];
  const float* l1f_whh = (const float*)d_in[4];
  const float* l1f_bih = (const float*)d_in[5];
  const float* l1f_bhh = (const float*)d_in[6];
  const float* l1b_wih = (const float*)d_in[7];
  const float* l1b_whh = (const float*)d_in[8];
  const float* l1b_bih = (const float*)d_in[9];
  const float* l1b_bhh = (const float*)d_in[10];
  const float* l2f_wih = (const float*)d_in[11];
  const float* l2f_whh = (const float*)d_in[12];
  const float* l2f_bih = (const float*)d_in[13];
  const float* l2f_bhh = (const float*)d_in[14];
  const float* l2b_wih = (const float*)d_in[15];
  const float* l2b_whh = (const float*)d_in[16];
  const float* l2b_bih = (const float*)d_in[17];
  const float* l2b_bhh = (const float*)d_in[18];
  const float* Wout  = (const float*)d_in[19];
  const float* bout  = (const float*)d_in[20];
  const float* trans = (const float*)d_in[21];
  float* out = (float*)d_out;

  unsigned char* ws = (unsigned char*)d_ws;
  size_t off = 0;
  auto take = [&](size_t bytes) -> void* {
    void* p = ws + off;
    off += (bytes + 255) & ~(size_t)255;
    return p;
  };

  bf16* xe     = (bf16*)take((size_t)SS * BB * EE * 2);        // embedded input (bf16)
  bf16* l1out  = (bf16*)take((size_t)SS * BB * 2 * HH * 2);    // layer1 output  (bf16)
  bf16* l2out  = (bf16*)take((size_t)SS * BB * 2 * HH * 2);    // layer2 output  (bf16)
  bf16* w1f_ih = (bf16*)take((size_t)4 * HH * EE * 2);
  bf16* w1f_hh = (bf16*)take((size_t)4 * HH * HH * 2);
  bf16* w1b_ih = (bf16*)take((size_t)4 * HH * EE * 2);
  bf16* w1b_hh = (bf16*)take((size_t)4 * HH * HH * 2);
  bf16* w2f_ih = (bf16*)take((size_t)4 * HH * 2 * HH * 2);
  bf16* w2f_hh = (bf16*)take((size_t)4 * HH * HH * 2);
  bf16* w2b_ih = (bf16*)take((size_t)4 * HH * 2 * HH * 2);
  bf16* w2b_hh = (bf16*)take((size_t)4 * HH * HH * 2);
  bf16*  hstate = (bf16*)take((size_t)2 * 2 * BB * HH * 2);    // [dir][buf][B][H]
  float* cstate = (float*)take((size_t)2 * BB * HH * 4);       // contiguous after hstate
  float* emis   = (float*)take((size_t)SS * BB * TT * 4);
  float* scoreb = (float*)take((size_t)BB * 4);
  float* partb  = (float*)take((size_t)BB * 4);

  // Convert weights fp32 -> bf16 (one-time per call; ~5 MB, L2-resident afterwards)
  auto cvt = [&](const float* s, bf16* d, int n) {
    cvt_bf16_kernel<<<(n + 255) / 256, 256, 0, stream>>>(s, d, n);
  };
  cvt(l1f_wih, w1f_ih, 4 * HH * EE);
  cvt(l1f_whh, w1f_hh, 4 * HH * HH);
  cvt(l1b_wih, w1b_ih, 4 * HH * EE);
  cvt(l1b_whh, w1b_hh, 4 * HH * HH);
  cvt(l2f_wih, w2f_ih, 4 * HH * 2 * HH);
  cvt(l2f_whh, w2f_hh, 4 * HH * HH);
  cvt(l2b_wih, w2b_ih, 4 * HH * 2 * HH);
  cvt(l2b_whh, w2b_hh, 4 * HH * HH);

  // Embedding gather
  embed_kernel<<<(SS * BB * (EE / 8) + 255) / 256, 256, 0, stream>>>(sent, embedw, xe);

  // hstate + cstate are contiguous (sizes are 256B multiples): zero both at once
  const size_t state_dw = ((size_t)2 * 2 * BB * HH * 2 + (size_t)2 * BB * HH * 4) / 4;

  // Layer 1 (IN_DIM = EE = 256)
  zero_kernel<<<256, 256, 0, stream>>>((uint32_t*)hstate, state_dw);
  for (int t = 0; t < SS; ++t) {
    lstm_step_kernel<EE><<<64, 256, 0, stream>>>(
        xe, w1f_ih, w1f_hh, w1b_ih, w1b_hh,
        l1f_bih, l1f_bhh, l1b_bih, l1b_bhh,
        hstate, cstate, l1out, t);
  }

  // Layer 2 (IN_DIM = 2*HH = 512)
  zero_kernel<<<256, 256, 0, stream>>>((uint32_t*)hstate, state_dw);
  for (int t = 0; t < SS; ++t) {
    lstm_step_kernel<2 * HH><<<64, 256, 0, stream>>>(
        l1out, w2f_ih, w2f_hh, w2b_ih, w2b_hh,
        l2f_bih, l2f_bhh, l2b_bih, l2b_bhh,
        hstate, cstate, l2out, t);
  }

  // Emissions + CRF
  emissions_kernel<<<(SS * BB * 32 + 255) / 256, 256, 0, stream>>>(l2out, Wout, bout, emis);
  gold_kernel<<<1, 256, 0, stream>>>(labels, emis, trans, scoreb);
  crf_forward_kernel<<<(BB * 32 + 255) / 256, 256, 0, stream>>>(emis, trans, partb);
  loss_kernel<<<1, 256, 0, stream>>>(partb, scoreb, out);
}
